// HyperEmbedding_38053410242731
// MI455X (gfx1250) — matrix operand order repaired and checked by reading, not compiled, not run on hardware
//
#include <hip/hip_runtime.h>
#include <hip/hip_bf16.h>

#define BB 4
#define CC 128
#define IMG 224
#define KP 16
#define WVD 128
#define EMB 768
#define RANK 64
#define NGRID 14
#define NPATCH 196
#define NPAD 208
#define PDIM 256
#define NTOK (BB*CC)      // 512
#define NV (RANK*PDIM)    // 16384
#define NU (EMB*RANK)     // 49152

typedef __attribute__((ext_vector_type(16))) __bf16 v16bf;
typedef __attribute__((ext_vector_type(8)))  __bf16 v8bf;
typedef __attribute__((ext_vector_type(8)))  float  v8f;

__device__ __forceinline__ float gelu_exact(float x) {
    return 0.5f * x * (1.0f + erff(x * 0.70710678118654752440f));
}

__device__ __forceinline__ v16bf cat8(v8bf lo, v8bf hi) {
    return __builtin_shufflevector(lo, hi, 0,1,2,3,4,5,6,7,8,9,10,11,12,13,14,15);
}

// LayerNorm over 128 values, one value per thread (blockDim.x == 128).
__device__ float block_ln128(float x, const float* __restrict__ g,
                             const float* __restrict__ b, float* s_red, int t) {
    s_red[t] = x; __syncthreads();
    for (int s = 64; s > 0; s >>= 1) { if (t < s) s_red[t] += s_red[t + s]; __syncthreads(); }
    float m = s_red[0] * (1.0f / 128.0f); __syncthreads();
    float xc = x - m;
    s_red[t] = xc * xc; __syncthreads();
    for (int s = 64; s > 0; s >>= 1) { if (t < s) s_red[t] += s_red[t + s]; __syncthreads(); }
    float v = s_red[0] * (1.0f / 128.0f); __syncthreads();
    return xc * rsqrtf(v + 1e-5f) * g[t] + b[t];
}

// ---------------- Kernel 1: per-patch avg/max pooling ----------------
__global__ void pool_kernel(const float* __restrict__ x,
                            float* __restrict__ avg, float* __restrict__ mx) {
    int bc = blockIdx.x;
    const float* xp = x + (size_t)bc * IMG * IMG;
    int wv = threadIdx.x >> 5, lane = threadIdx.x & 31;
    for (int n = wv; n < NPATCH; n += 8) {
        int gy = n / NGRID, gx = n % NGRID;
        const float* base = xp + (size_t)(gy * KP) * IMG + gx * KP;
        float s = 0.0f, m = -3.4e38f;
        #pragma unroll
        for (int i = 0; i < 8; ++i) {
            int idx = lane + 32 * i;
            float v = base[(idx >> 4) * IMG + (idx & 15)];
            s += v; m = fmaxf(m, v);
        }
        #pragma unroll
        for (int off = 16; off > 0; off >>= 1) {
            s += __shfl_xor(s, off, 32);
            m = fmaxf(m, __shfl_xor(m, off, 32));
        }
        if (lane == 0) {
            avg[(size_t)bc * NPATCH + n] = s * (1.0f / 256.0f);
            mx[(size_t)bc * NPATCH + n]  = m;
        }
    }
}

// ---------------- Kernel 2: per-token MLP chain -> feat, hu, hv ----------------
__global__ void token_kernel(
    const float* __restrict__ avg, const float* __restrict__ mx, const float* __restrict__ meta,
    const float* __restrict__ aw1, const float* __restrict__ ab1,
    const float* __restrict__ aw2, const float* __restrict__ ab2,
    const float* __restrict__ mw1, const float* __restrict__ mb1,
    const float* __restrict__ mw2, const float* __restrict__ mb2,
    const float* __restrict__ fw1, const float* __restrict__ fb1,
    const float* __restrict__ fw2, const float* __restrict__ fb2,
    const float* __restrict__ lng, const float* __restrict__ lnb,
    const float* __restrict__ p1w, const float* __restrict__ p1b,
    const float* __restrict__ p2w, const float* __restrict__ p2b,
    const float* __restrict__ cfw, const float* __restrict__ cfb,
    const float* __restrict__ clng, const float* __restrict__ clnb,
    const float* __restrict__ ulng, const float* __restrict__ ulnb,
    const float* __restrict__ uw1, const float* __restrict__ ub1,
    const float* __restrict__ uw2, const float* __restrict__ ub2,
    const float* __restrict__ vlng, const float* __restrict__ vlnb,
    const float* __restrict__ vw1, const float* __restrict__ vb1,
    const float* __restrict__ vw2, const float* __restrict__ vb2,
    float* __restrict__ feat, float* __restrict__ hu, float* __restrict__ hv) {

    __shared__ float s_avg[NPATCH], s_mx[NPATCH];
    __shared__ float s_h[64], s_a[256], s_b[256], s_c[128], s_meta[128], s_red[128];
    int tok = blockIdx.x, t = threadIdx.x;

    for (int i = t; i < NPATCH; i += 128) {
        s_avg[i] = avg[(size_t)tok * NPATCH + i];
        s_mx[i]  = mx[(size_t)tok * NPATCH + i];
    }
    s_meta[t] = meta[(size_t)tok * WVD + t];
    __syncthreads();

    if (t < 49) {
        float acc = ab1[t];
        for (int p = 0; p < NPATCH; ++p) acc += s_avg[p] * aw1[t * NPATCH + p];
        s_h[t] = gelu_exact(acc);
    }
    __syncthreads();
    { float acc = ab2[t]; for (int p = 0; p < 49; ++p) acc += s_h[p] * aw2[t * 49 + p]; s_a[t] = acc; }
    __syncthreads();
    if (t < 49) {
        float acc = mb1[t];
        for (int p = 0; p < NPATCH; ++p) acc += s_mx[p] * mw1[t * NPATCH + p];
        s_h[t] = gelu_exact(acc);
    }
    __syncthreads();
    { float acc = mb2[t]; for (int p = 0; p < 49; ++p) acc += s_h[p] * mw2[t * 49 + p]; s_a[128 + t] = acc; }
    __syncthreads();
    #pragma unroll
    for (int rep = 0; rep < 2; ++rep) {
        int o = t + rep * 128;
        float acc = fb1[o];
        for (int p = 0; p < 256; ++p) acc += s_a[p] * fw1[o * 256 + p];
        s_b[o] = gelu_exact(acc);
    }
    __syncthreads();
    { float acc = fb2[t]; for (int p = 0; p < 256; ++p) acc += s_b[p] * fw2[t * 256 + p]; s_c[t] = acc; }
    __syncthreads();
    float content = block_ln128(s_c[t], lng, lnb, s_red, t);
    s_c[t] = content;
    __syncthreads();

    { float acc = p1b[t]; for (int p = 0; p < 128; ++p) acc += s_meta[p] * p1w[t * 128 + p]; s_a[t] = acc; }
    { float acc = p2b[t]; for (int p = 0; p < 128; ++p) acc += s_c[p]   * p2w[t * 128 + p]; s_a[128 + t] = acc; }
    __syncthreads();
    float pre;
    { float acc = cfb[t]; for (int p = 0; p < 256; ++p) acc += s_a[p] * cfw[t * 256 + p]; pre = acc; }
    float cm = gelu_exact(block_ln128(pre, clng, clnb, s_red, t)) + s_a[t];
    float ft = cm + s_meta[t];
    s_c[t] = ft;
    feat[(size_t)tok * WVD + t] = ft;
    __syncthreads();

    float lnu = block_ln128(s_c[t], ulng, ulnb, s_red, t);
    s_b[t] = lnu;
    __syncthreads();
    { float acc = ub1[t]; for (int p = 0; p < 128; ++p) acc += s_b[p] * uw1[t * 128 + p]; s_a[t] = gelu_exact(acc); }
    __syncthreads();
    if (t < 64) {
        float acc = ub2[t];
        for (int p = 0; p < 128; ++p) acc += s_a[p] * uw2[t * 128 + p];
        hu[(size_t)tok * RANK + t] = gelu_exact(acc);
    }
    __syncthreads();
    float lnv = block_ln128(s_c[t], vlng, vlnb, s_red, t);
    s_b[t] = lnv;
    __syncthreads();
    { float acc = vb1[t]; for (int p = 0; p < 128; ++p) acc += s_b[p] * vw1[t * 128 + p]; s_a[t] = gelu_exact(acc); }
    __syncthreads();
    if (t < 64) {
        float acc = vb2[t];
        for (int p = 0; p < 128; ++p) acc += s_a[p] * vw2[t * 128 + p];
        hv[(size_t)tok * RANK + t] = gelu_exact(acc);
    }
}

// ---------------- Kernel 3: bias head ----------------
__global__ void bias_kernel(const float* __restrict__ feat,
                            const float* __restrict__ blng, const float* __restrict__ blnb,
                            const float* __restrict__ bw1, const float* __restrict__ bb1,
                            const float* __restrict__ bw2, const float* __restrict__ bb2,
                            float* __restrict__ biasv) {
    __shared__ float s_sf[128], s_h[64], s_red[128];
    int b = blockIdx.x, t = threadIdx.x;
    float s = 0.0f;
    for (int c = 0; c < CC; ++c) s += feat[((size_t)(b * CC + c)) * WVD + t];
    float sf = s * (1.0f / 128.0f);
    s_sf[t] = block_ln128(sf, blng, blnb, s_red, t);
    __syncthreads();
    if (t < 64) {
        float acc = bb1[t];
        for (int p = 0; p < 128; ++p) acc += s_sf[p] * bw1[t * 128 + p];
        s_h[t] = gelu_exact(acc);
    }
    __syncthreads();
    for (int e = t; e < EMB; e += 128) {
        float acc = bb2[e];
        for (int p = 0; p < 64; ++p) acc += s_h[p] * bw2[e * 64 + p];
        biasv[(size_t)b * EMB + e] = acc;
    }
}

// ---------------- Kernel 4: hypernet GEMM  out[512,N] = H[512,64] @ W[N,64]^T + bias ----------------
// 256 threads = 8 waves, each a 16x16 tile; LDS-repacked coalesced b128 stores.
__global__ void hyper_gemm(const float* __restrict__ H, const float* __restrict__ W,
                           const float* __restrict__ bias, __bf16* __restrict__ out, int N) {
    __shared__ __bf16 s_out[16][128];
    int m0 = blockIdx.x * 16;
    int wv = threadIdx.x >> 5, lane = threadIdx.x & 31;
    int h = lane >> 4, m = lane & 15;
    int n0 = blockIdx.y * 128 + wv * 16;
    v8f acc = {};
    #pragma unroll
    for (int k0 = 0; k0 < RANK; k0 += 32) {
        v16bf a, bf;
        const float* Arow = H + (size_t)(m0 + m) * RANK + k0;
        const float* Brow = W + (size_t)(n0 + m) * RANK + k0 + 16 * h;
        #pragma unroll
        for (int e = 0; e < 16; ++e) {
            int ka = e + 16 * h;                          // A 16x32: K = e + 16h, M = m
            a[e]  = (__bf16)Arow[ka];
            bf[e] = (__bf16)Brow[e];                      // B 32x16: K = e + 16h, N = m
        }
        acc = __builtin_amdgcn_wmma_f32_16x16x32_bf16(false, a, false, bf, (short)0, acc, false, false);
    }
    float bn = bias[n0 + m];
    #pragma unroll
    for (int d = 0; d < 8; ++d)
        s_out[d + 8 * h][wv * 16 + m] = (__bf16)(acc[d] + bn);
    __syncthreads();
    int r = threadIdx.x >> 4, chk = threadIdx.x & 15;     // 16 rows x 16 chunks of 8 bf16
    *(uint4*)(out + (size_t)(m0 + r) * N + (size_t)blockIdx.y * 128 + chk * 8) =
        *(const uint4*)(&s_out[r][chk * 8]);
}

// ---------------- Kernel 5: inter[bc, n(208), r(64)] = patches[bc][196,256] @ V[bc]^T ----------------
// 128 threads = 4 waves; grid = (NTOK, 13); LDS-repacked coalesced stores.
__global__ void inter_gemm(const float* __restrict__ x, const __bf16* __restrict__ V,
                           __bf16* __restrict__ inter) {
    __shared__ __bf16 As[16][32];
    __shared__ __bf16 s_out[16][64];
    int bc = blockIdx.x;
    int m0 = blockIdx.y * 16;
    int wv = threadIdx.x >> 5, lane = threadIdx.x & 31;
    int h = lane >> 4, m = lane & 15;
    int n0 = wv * 16;
    const float*  xp = x + (size_t)bc * IMG * IMG;
    const __bf16* Vp = V + (size_t)bc * NV;
    v8f acc = {};
    for (int k0 = 0; k0 < PDIM; k0 += 32) {
        #pragma unroll
        for (int i = threadIdx.x; i < 512; i += 128) {
            int row = i >> 5, kk = i & 31;
            int n = m0 + row;
            float v = 0.0f;
            if (n < NPATCH) {
                int p = k0 + kk;
                int gy = n / NGRID, gx = n % NGRID;
                v = xp[(size_t)(gy * KP + (p >> 4)) * IMG + gx * KP + (p & 15)];
            }
            As[row][kk] = (__bf16)v;
        }
        __syncthreads();
        v16bf a = *(const v16bf*)(&As[m][16 * h]);
        v16bf bf = *(const v16bf*)(Vp + (size_t)(n0 + m) * PDIM + k0 + 16 * h);
        acc = __builtin_amdgcn_wmma_f32_16x16x32_bf16(false, a, false, bf, (short)0, acc, false, false);
        __syncthreads();
    }
    #pragma unroll
    for (int d = 0; d < 8; ++d)
        s_out[d + 8 * h][n0 + m] = (__bf16)acc[d];       // padded rows (196..207) hold zeros
    __syncthreads();
    int r = threadIdx.x >> 3, chk = threadIdx.x & 7;      // 16 rows x 8 chunks of 8 bf16
    *(uint4*)(inter + ((size_t)bc * NPAD + m0 + r) * RANK + chk * 8) =
        *(const uint4*)(&s_out[r][chk * 8]);
}

// ---------------- Kernel 6: out[b,n,e] = (sum_c inter @ U^T + bias) * scaler ----------------
// 256 threads = 8 waves; each wave owns TWO adjacent 16-wide e-tiles (shared A fragment,
// two independent WMMA chains). grid = (BB, 13, EMB/256 = 3). K = C*RANK = 8192.
__global__ void out_gemm(const __bf16* __restrict__ inter, const __bf16* __restrict__ U,
                         const float* __restrict__ biasv, const float* __restrict__ scaler,
                         float* __restrict__ out) {
    int b = blockIdx.x;
    int m0 = blockIdx.y * 16;
    int wv = threadIdx.x >> 5, lane = threadIdx.x & 31;
    int h = lane >> 4, m = lane & 15;
    int e0 = blockIdx.z * 256 + wv * 32;
    v8f acc0 = {}, acc1 = {};
    for (int c = 0; c < CC; ++c) {
        const __bf16* Ap  = inter + ((size_t)(b * CC + c) * NPAD + m0 + m) * RANK;
        const __bf16* Bp0 = U + (size_t)(b * CC + c) * NU + (size_t)(e0 + m) * RANK;
        const __bf16* Bp1 = Bp0 + (size_t)16 * RANK;
        #pragma unroll
        for (int k0 = 0; k0 < RANK; k0 += 32) {
            v16bf a  = *(const v16bf*)(Ap + k0 + 16 * h);
            v16bf b0 = *(const v16bf*)(Bp0 + k0 + 16 * h);
            v16bf b1 = *(const v16bf*)(Bp1 + k0 + 16 * h);
            acc0 = __builtin_amdgcn_wmma_f32_16x16x32_bf16(false, a, false, b0, (short)0, acc0, false, false);
            acc1 = __builtin_amdgcn_wmma_f32_16x16x32_bf16(false, a, false, b1, (short)0, acc1, false, false);
        }
    }
    float sc = *scaler;
    #pragma unroll
    for (int d = 0; d < 8; ++d) {
        int n = m0 + d + 8 * h;
        if (n < NPATCH) {
            int e = e0 + m;
            out[((size_t)b * NPATCH + n) * EMB + e]      = (acc0[d] + biasv[(size_t)b * EMB + e])      * sc;
            out[((size_t)b * NPATCH + n) * EMB + e + 16] = (acc1[d] + biasv[(size_t)b * EMB + e + 16]) * sc;
        }
    }
}

extern "C" void kernel_launch(void* const* d_in, const int* in_sizes, int n_in,
                              void* d_out, int out_size, void* d_ws, size_t ws_size,
                              hipStream_t stream) {
    const float* x    = (const float*)d_in[0];
    const float* meta = (const float*)d_in[1];
    const float* aw1 = (const float*)d_in[2];  const float* ab1 = (const float*)d_in[3];
    const float* aw2 = (const float*)d_in[4];  const float* ab2 = (const float*)d_in[5];
    const float* mw1 = (const float*)d_in[6];  const float* mb1 = (const float*)d_in[7];
    const float* mw2 = (const float*)d_in[8];  const float* mb2 = (const float*)d_in[9];
    const float* fw1 = (const float*)d_in[10]; const float* fb1 = (const float*)d_in[11];
    const float* fw2 = (const float*)d_in[12]; const float* fb2 = (const float*)d_in[13];
    const float* lng = (const float*)d_in[14]; const float* lnb = (const float*)d_in[15];
    const float* p1w = (const float*)d_in[16]; const float* p1b = (const float*)d_in[17];
    const float* p2w = (const float*)d_in[18]; const float* p2b = (const float*)d_in[19];
    const float* cfw = (const float*)d_in[20]; const float* cfb = (const float*)d_in[21];
    const float* clng = (const float*)d_in[22]; const float* clnb = (const float*)d_in[23];
    const float* ulng = (const float*)d_in[24]; const float* ulnb = (const float*)d_in[25];
    const float* uw1 = (const float*)d_in[26]; const float* ub1 = (const float*)d_in[27];
    const float* uw2 = (const float*)d_in[28]; const float* ub2 = (const float*)d_in[29];
    const float* uw3 = (const float*)d_in[30]; const float* ub3 = (const float*)d_in[31];
    const float* vlng = (const float*)d_in[32]; const float* vlnb = (const float*)d_in[33];
    const float* vw1 = (const float*)d_in[34]; const float* vb1 = (const float*)d_in[35];
    const float* vw2 = (const float*)d_in[36]; const float* vb2 = (const float*)d_in[37];
    const float* vw3 = (const float*)d_in[38]; const float* vb3 = (const float*)d_in[39];
    const float* blng = (const float*)d_in[40]; const float* blnb = (const float*)d_in[41];
    const float* bw1 = (const float*)d_in[42]; const float* bb1 = (const float*)d_in[43];
    const float* bw2 = (const float*)d_in[44]; const float* bb2 = (const float*)d_in[45];
    const float* scaler = (const float*)d_in[46];

    char* wsb = (char*)d_ws;
    size_t off = 0;
    auto carve = [&](size_t bytes) { void* p = wsb + off; off += (bytes + 255) & ~(size_t)255; return p; };
    float*  avg   = (float*)carve((size_t)NTOK * NPATCH * 4);
    float*  mx    = (float*)carve((size_t)NTOK * NPATCH * 4);
    float*  feat  = (float*)carve((size_t)NTOK * WVD * 4);
    float*  hu    = (float*)carve((size_t)NTOK * RANK * 4);
    float*  hv    = (float*)carve((size_t)NTOK * RANK * 4);
    float*  biasv = (float*)carve((size_t)BB * EMB * 4);
    __bf16* Vb    = (__bf16*)carve((size_t)NTOK * NV * 2);
    __bf16* Ub    = (__bf16*)carve((size_t)NTOK * NU * 2);
    __bf16* inter = (__bf16*)carve((size_t)NTOK * NPAD * RANK * 2);
    float* out = (float*)d_out;

    pool_kernel<<<NTOK, 256, 0, stream>>>(x, avg, mx);
    token_kernel<<<NTOK, 128, 0, stream>>>(avg, mx, meta,
        aw1, ab1, aw2, ab2, mw1, mb1, mw2, mb2, fw1, fb1, fw2, fb2, lng, lnb,
        p1w, p1b, p2w, p2b, cfw, cfb, clng, clnb,
        ulng, ulnb, uw1, ub1, uw2, ub2, vlng, vlnb, vw1, vb1, vw2, vb2,
        feat, hu, hv);
    bias_kernel<<<BB, 128, 0, stream>>>(feat, blng, blnb, bw1, bb1, bw2, bb2, biasv);
    hyper_gemm<<<dim3(NTOK / 16, NV / 128), 256, 0, stream>>>(hv, vw3, vb3, Vb, NV);
    hyper_gemm<<<dim3(NTOK / 16, NU / 128), 256, 0, stream>>>(hu, uw3, ub3, Ub, NU);
    inter_gemm<<<dim3(NTOK, NPAD / 16), 128, 0, stream>>>(x, Vb, inter);
    out_gemm<<<dim3(BB, NPAD / 16, EMB / 256), 256, 0, stream>>>(inter, Ub, biasv, scaler, out);
}
